// VisionTransformerPredictorAC_28647431865009
// MI455X (gfx1250) — matrix operand order repaired and checked by reading, not compiled, not run on hardware
//
#include <hip/hip_runtime.h>
#include <hip/hip_bf16.h>

// ---------------------------------------------------------------------------
// MI455X (gfx1250) implementation of the 4-layer ViT predictor.
// - All contractions: v_wmma_f32_16x16x32_f16 (wave32), f32 accumulate.
// - Weight tiles double-buffered in LDS via GLOBAL_LOAD_ASYNC_TO_LDS_B128
//   (ASYNCcnt); activations stream through the 192MB L2.
// - Fragment conversion is kept register-disjoint from the WMMA chain to
//   avoid the WMMA->VALU WAR hazard nops (4 co-exec slots for f16 WMMA).
// - Attention exploits the prefix-causal mask to prune ~2x of the work.
// ---------------------------------------------------------------------------

typedef __attribute__((ext_vector_type(16))) _Float16 v16h;
typedef __attribute__((ext_vector_type(8)))  _Float16 v8h;
typedef __attribute__((ext_vector_type(8)))  float    v8f;

union H16 { v16h v; v8h h[2]; };

#define WMMA_F16(A, B, C) \
  __builtin_amdgcn_wmma_f32_16x16x32_f16(false, (A), false, (B), (short)0, (C), false, false)

// Model constants
#define SEQ_N    1584
#define N_TEXT   16
#define F_TOK    1568
#define DIM      1024
#define HEADS    16
#define HD       64
#define QKV_N    3072
#define MLP_H    4096
#define EMBED    768
#define TXT_DIM  4096
#define SCALE_QK 0.125f
#define SC_COLS  1600       // 1584 padded to a multiple of 32

// ---- gfx1250 async global->LDS copy (ASYNCcnt) ----------------------------
__device__ __forceinline__ void async_copy_b128(void* lds_dst, const void* gsrc) {
  asm volatile("global_load_async_to_lds_b128 %0, %1, off"
               :: "v"((uint32_t)(uintptr_t)lds_dst),
                  "v"((uint64_t)(uintptr_t)gsrc)
               : "memory");
}
__device__ __forceinline__ void wait_async0() {
  asm volatile("s_wait_asynccnt 0" ::: "memory");
}

// 16 raw f32 of one A fragment (two 8-float runs), kept as float4s so the
// global loads vectorize to global_load_b128.
struct AF { float4 x0, x1, x2, x3; };

__device__ __forceinline__ void load_af(AF& r, const float* p) {
  r.x0 = *(const float4*)(p);
  r.x1 = *(const float4*)(p + 4);
  r.x2 = *(const float4*)(p + 16);
  r.x3 = *(const float4*)(p + 20);
}
__device__ __forceinline__ void cvt_af(H16& a, const AF& r) {
  a.v[0]  = (_Float16)r.x0.x; a.v[1]  = (_Float16)r.x0.y;
  a.v[2]  = (_Float16)r.x0.z; a.v[3]  = (_Float16)r.x0.w;
  a.v[4]  = (_Float16)r.x1.x; a.v[5]  = (_Float16)r.x1.y;
  a.v[6]  = (_Float16)r.x1.z; a.v[7]  = (_Float16)r.x1.w;
  a.v[8]  = (_Float16)r.x2.x; a.v[9]  = (_Float16)r.x2.y;
  a.v[10] = (_Float16)r.x2.z; a.v[11] = (_Float16)r.x2.w;
  a.v[12] = (_Float16)r.x3.x; a.v[13] = (_Float16)r.x3.y;
  a.v[14] = (_Float16)r.x3.z; a.v[15] = (_Float16)r.x3.w;
}

// ---------------------------------------------------------------------------
// Tiled GEMM:  C[M,N] = epi( A[M,K] @ B[K,N] + bias ) (+resid)
// Block tile 128x128, BK=32. 8 waves; wave (wm,wn) computes 32x64 (8 WMMAs
// per K-step). B tile: async double-buffered f32 in LDS (32KB). A fragments:
// direct from global (L2-resident), f32->f16 in flight.
// Requires K % 32 == 0 and N % 128 == 0 (true for every GEMM here).
// act: 0 = none, 1 = exact GELU. resid may be null.
// ---------------------------------------------------------------------------
__global__ __launch_bounds__(256) void gemm_kernel(
    const float* __restrict__ A, const float* __restrict__ B,
    const float* __restrict__ bias, const float* __restrict__ resid,
    float* __restrict__ C, int M, int N, int K, int act)
{
  __shared__ __align__(16) float Bst[2][32 * 128];

  const int tid  = threadIdx.x;
  const int bn0  = blockIdx.x * 128;
  const int bm0  = blockIdx.y * 128;
  const int lane = tid & 31;
  const int wave = tid >> 5;
  const int wm   = wave >> 1;          // 0..3 -> 32-row sub-tile
  const int wn   = wave & 1;           // 0..1 -> 64-col sub-tile

  // cooperative async-stage coordinates for the B tile (32x128 f32)
  const int brow = tid >> 3;           // 0..31
  const int bcol = (tid & 7) * 16;     // 0..112
  const float* Bbase = B + (size_t)brow * N + bn0 + bcol;

  // A fragment rows (ISA A-layout: lane m = L&15, dual 8-runs at kb, kb+16)
  const int m  = lane & 15;
  const int kb = (lane >> 4) * 8;
  int r0 = bm0 + wm * 32 + m;      if (r0 > M - 1) r0 = M - 1;
  int r1 = bm0 + wm * 32 + 16 + m; if (r1 > M - 1) r1 = M - 1;
  const float* A0 = A + (size_t)r0 * K + kb;
  const float* A1 = A + (size_t)r1 * K + kb;

  v8f acc[2][4];
#pragma unroll
  for (int s = 0; s < 2; ++s)
#pragma unroll
    for (int j = 0; j < 4; ++j) acc[s][j] = v8f{};

  const int nT = K >> 5;

  // prologue: async-stage B tile 0 into buffer 0
#pragma unroll
  for (int ch = 0; ch < 4; ++ch)
    async_copy_b128(&Bst[0][brow * 128 + bcol + ch * 4], Bbase + ch * 4);
  wait_async0();
  __syncthreads();

  for (int t = 0; t < nT; ++t) {
    const int cur = t & 1;
    const int k0  = t << 5;

    if (t + 1 < nT) {   // issue next B tile while computing this one
      const float* Bn = Bbase + (size_t)(k0 + 32) * N;
#pragma unroll
      for (int ch = 0; ch < 4; ++ch)
        async_copy_b128(&Bst[cur ^ 1][brow * 128 + bcol + ch * 4], Bn + ch * 4);
      __builtin_prefetch(A0 + k0 + 32, 0, 1);
      __builtin_prefetch(A1 + k0 + 32, 0, 1);
    }

    // 1) issue A raw loads (L2 hits) first so the wait lands after the
    //    B-conversion work below.
    AF ra0, ra1;
    load_af(ra0, A0 + k0);
    load_af(ra1, A1 + k0);

    // 2) convert all 4 B fragments into distinct registers (no WAR with the
    //    WMMA chain below -> no hazard nops).
    const float* bp = &Bst[cur][wn * 64 + (lane & 15)];
    v16h bf[4];
#pragma unroll
    for (int j = 0; j < 4; ++j) {
      const float* bj = bp + j * 16;
#pragma unroll
      for (int i = 0; i < 16; ++i) bf[j][i] = (_Float16)bj[(kb + (i & 7) + (i >> 3) * 16) * 128];
    }

    // 3) convert A fragments (loads have completed under the B work).
    H16 a0, a1;
    cvt_af(a0, ra0);
    cvt_af(a1, ra1);

    // 4) 8 back-to-back WMMAs (operands read-only, accumulators chained).
#pragma unroll
    for (int j = 0; j < 4; ++j) {
      acc[0][j] = WMMA_F16(a0.v, bf[j], acc[0][j]);
      acc[1][j] = WMMA_F16(a1.v, bf[j], acc[1][j]);
    }

    if (t + 1 < nT) wait_async0();
    __syncthreads();
  }

  // epilogue: C/D layout -> VGPR r = row r (lanes 0-15) / r+8 (lanes 16-31)
  const int m8 = (lane >> 4) * 8;
  const int n  = lane & 15;
#pragma unroll
  for (int s = 0; s < 2; ++s) {
#pragma unroll
    for (int j = 0; j < 4; ++j) {
      const int col = bn0 + wn * 64 + j * 16 + n;
      const float bv = bias ? bias[col] : 0.f;
#pragma unroll
      for (int r = 0; r < 8; ++r) {
        const int row = bm0 + wm * 32 + s * 16 + m8 + r;
        if (row < M) {
          float v = acc[s][j][r] + bv;
          if (act) v = 0.5f * v * (1.f + erff(v * 0.70710678118654752f));
          const size_t idx = (size_t)row * N + col;
          if (resid) v += resid[idx];
          C[idx] = v;
        }
      }
    }
  }
}

// ---------------------------------------------------------------------------
// LayerNorm: one block per row, D = 1024.
// ---------------------------------------------------------------------------
__global__ __launch_bounds__(256) void ln_kernel(
    const float* __restrict__ X, const float* __restrict__ w,
    const float* __restrict__ b, float* __restrict__ Y, int D)
{
  const int row = blockIdx.x;
  const float* x = X + (size_t)row * D;
  float* y = Y + (size_t)row * D;
  const int tid = threadIdx.x;

  float s = 0.f, ss = 0.f;
  for (int i = tid; i < D; i += 256) { float v = x[i]; s += v; ss += v * v; }

  __shared__ float r1[256], r2[256];
  r1[tid] = s; r2[tid] = ss;
  __syncthreads();
  for (int st = 128; st > 0; st >>= 1) {
    if (tid < st) { r1[tid] += r1[tid + st]; r2[tid] += r2[tid + st]; }
    __syncthreads();
  }
  const float mean = r1[0] / (float)D;
  const float var  = r2[0] / (float)D - mean * mean;
  const float inv  = rsqrtf(var + 1e-5f);
  for (int i = tid; i < D; i += 256)
    y[i] = (x[i] - mean) * inv * w[i] + b[i];
}

// ---------------------------------------------------------------------------
// RoPE on Q,K of qkv, in place. grid.x = 1568 frame tokens (text = identity).
// pairs 0-9: t axis, 10-19: h, 20-29: w (inv freq 10000^{-i/10}), 30-31 id.
// ---------------------------------------------------------------------------
__global__ __launch_bounds__(256) void rope_kernel(float* __restrict__ qkv)
{
  const int ft    = blockIdx.x;
  const int token = N_TEXT + ft;
  const float p0 = (float)(ft / 196);
  const float p1 = (float)((ft % 196) / 14);
  const float p2 = (float)(ft % 14);

  const int tid = threadIdx.x;
#pragma unroll
  for (int it = 0; it < 2; ++it) {
    const int wi = tid + it * 256;      // 0..511 = 16 heads x 32 pairs
    const int h = wi >> 5;
    const int p = wi & 31;
    if (p < 30) {
      const int axis = p / 10;
      const int i    = p % 10;
      const float pos = (axis == 0) ? p0 : ((axis == 1) ? p1 : p2);
      const float ang = pos * powf(10000.f, -(float)i * 0.1f);
      float s, c;
      sincosf(ang, &s, &c);
      float* qp = qkv + (size_t)token * QKV_N + h * HD + 2 * p;
      float* kp = qp + DIM;
      const float q0 = qp[0], q1 = qp[1];
      qp[0] = q0 * c - q1 * s;
      qp[1] = q1 * c + q0 * s;
      const float k0 = kp[0], k1 = kp[1];
      kp[0] = k0 * c - k1 * s;
      kp[1] = k1 * c + k0 * s;
    }
  }
}

// Prefix-causal column limit for query row q: attends exactly cols [0,clim).
__device__ __forceinline__ int row_limit(int q)
{
  return (q < N_TEXT) ? (q + 1) : (N_TEXT + ((q - N_TEXT) / 196 + 1) * 196);
}

// ---------------------------------------------------------------------------
// Fused attention: one wave per (16-row q-tile, head).
// QK^T via WMMA -> f16 strip in LDS, pruned masked softmax, P@V via WMMA.
// Work pruned to the tile's causal limit (~2x savings on average).
// LDS: 16x1600 f16 scores + Q(16x64) + Kt(64x16) f16 = 55KB.
// ---------------------------------------------------------------------------
__global__ __launch_bounds__(32) void attn_kernel(
    const float* __restrict__ qkv, float* __restrict__ obuf)
{
  const int qt   = blockIdx.x;     // 0..98
  const int h    = blockIdx.y;     // 0..15
  const int lane = threadIdx.x;

  __shared__ __align__(32) _Float16 qsh[16 * HD];
  __shared__ __align__(32) _Float16 kts[HD * 16];
  __shared__ __align__(32) _Float16 sc[16 * SC_COLS];

  const int clim_tile = row_limit(qt * 16 + 15);   // max prefix in this tile
  const int ktmax     = (clim_tile + 15) >> 4;     // 16-col score tiles
  const int nch       = (clim_tile + 31) >> 5;     // 32-token P@V chunks

  // ---- load Q tile (16 x 64) ----
  {
    const int r = lane >> 1, c0 = (lane & 1) * 32;
    const float* qp = qkv + (size_t)(qt * 16 + r) * QKV_N + h * HD + c0;
#pragma unroll
    for (int i = 0; i < 32; ++i) qsh[r * HD + c0 + i] = (_Float16)qp[i];
  }
  __syncthreads();

  const int m  = lane & 15;
  const int kb = (lane >> 4) * 8;

  H16 aq0, aq1;
  aq0.h[0] = *(const v8h*)&qsh[m * HD + kb];
  aq0.h[1] = *(const v8h*)&qsh[m * HD + kb + 16];
  aq1.h[0] = *(const v8h*)&qsh[m * HD + 32 + kb];
  aq1.h[1] = *(const v8h*)&qsh[m * HD + 32 + kb + 16];

  // ---- scores: S = Q K^T * scale, only kt < ktmax ----
  for (int kt = 0; kt < ktmax; ++kt) {
    {   // stage K tile transposed: kts[d][token]
      const int r = lane >> 1, c0 = (lane & 1) * 32;
      const float* kp = qkv + (size_t)(kt * 16 + r) * QKV_N + DIM + h * HD + c0;
#pragma unroll
      for (int i = 0; i < 32; ++i) kts[(c0 + i) * 16 + r] = (_Float16)kp[i];
    }
    __syncthreads();

    v16h b0, b1;
#pragma unroll
    for (int i = 0; i < 16; ++i) {
      const int kd = kb + (i & 7) + (i >> 3) * 16;
      b0[i] = kts[kd * 16 + (lane & 15)];
      b1[i] = kts[(32 + kd) * 16 + (lane & 15)];
    }
    v8f acc = v8f{};
    acc = WMMA_F16(aq0.v, b0, acc);
    acc = WMMA_F16(aq1.v, b1, acc);

    const int m8 = (lane >> 4) * 8;
    const int n  = lane & 15;
#pragma unroll
    for (int r = 0; r < 8; ++r)
      sc[(m8 + r) * SC_COLS + kt * 16 + n] = (_Float16)(acc[r] * SCALE_QK);
    __syncthreads();
  }

  // ---- softmax: 2 lanes per row (halves combined via shfl_xor 16) ----
  {
    const int row  = lane & 15;
    const int half = lane >> 4;
    const int qg   = qt * 16 + row;
    const int clim = row_limit(qg);
    _Float16* rp = &sc[row * SC_COLS];
    const int lo = half * 792;                        // 792*2 = 1584
    int hi = lo + 792; if (clim < hi) hi = (clim > lo) ? clim : lo;

    float mx = -3.0e38f;
    for (int c = lo; c < hi; ++c) mx = fmaxf(mx, (float)rp[c]);
    mx = fmaxf(mx, __shfl_xor(mx, 16));

    float sum = 0.f;
    for (int c = lo; c < hi; ++c) {
      const float e = __expf((float)rp[c] - mx);
      sum += e;
      rp[c] = (_Float16)e;
    }
    sum += __shfl_xor(sum, 16);

    const float inv = 1.f / sum;
    for (int c = lo; c < hi; ++c) rp[c] = (_Float16)((float)rp[c] * inv);
    for (int c = hi; c < lo + 792; ++c) rp[c] = (_Float16)0.f;   // masked
    if (half)
      for (int c = SEQ_N; c < SC_COLS; ++c) rp[c] = (_Float16)0.f; // pad
  }
  __syncthreads();

  // ---- O = P @ V over pruned chunk range ----
  v8f o0 = v8f{}, o1 = v8f{}, o2 = v8f{}, o3 = v8f{};
  for (int c = 0; c < nch; ++c) {
    H16 ap;
    ap.h[0] = *(const v8h*)&sc[m * SC_COLS + c * 32 + kb];
    ap.h[1] = *(const v8h*)&sc[m * SC_COLS + c * 32 + kb + 16];

    v16h b0 = {}, b1 = {}, b2 = {}, b3 = {};
#pragma unroll
    for (int i = 0; i < 16; ++i) {
      const int t = c * 32 + kb + (i & 7) + (i >> 3) * 16;
      if (t < SEQ_N) {
        const float* vp = qkv + (size_t)t * QKV_N + 2 * DIM + h * HD + (lane & 15);
        b0[i] = (_Float16)vp[0];
        b1[i] = (_Float16)vp[16];
        b2[i] = (_Float16)vp[32];
        b3[i] = (_Float16)vp[48];
      }
    }
    o0 = WMMA_F16(ap.v, b0, o0);
    o1 = WMMA_F16(ap.v, b1, o1);
    o2 = WMMA_F16(ap.v, b2, o2);
    o3 = WMMA_F16(ap.v, b3, o3);
  }

  // ---- write O tile ----
  const int m8 = (lane >> 4) * 8;
  const int n  = lane & 15;
#pragma unroll
  for (int r = 0; r < 8; ++r) {
    const int token = qt * 16 + m8 + r;
    float* op = obuf + (size_t)token * DIM + h * HD;
    op[n]      = o0[r];
    op[16 + n] = o1[r];
    op[32 + n] = o2[r];
    op[48 + n] = o3[r];
  }
}

// ---------------------------------------------------------------------------
// Host-side orchestration
// ---------------------------------------------------------------------------
extern "C" void kernel_launch(void* const* d_in, const int* in_sizes, int n_in,
                              void* d_out, int out_size, void* d_ws, size_t ws_size,
                              hipStream_t stream)
{
  (void)in_sizes; (void)n_in; (void)out_size; (void)ws_size;

  const float* x        = (const float*)d_in[0];
  const float* text     = (const float*)d_in[1];
  const float* w_embed  = (const float*)d_in[2];
  const float* b_embed  = (const float*)d_in[3];
  const float* w_text   = (const float*)d_in[4];
  const float* b_text   = (const float*)d_in[5];
  const float* ln1_w    = (const float*)d_in[6];
  const float* ln1_b    = (const float*)d_in[7];
  const float* qkv_w    = (const float*)d_in[8];
  const float* qkv_b    = (const float*)d_in[9];
  const float* proj_w   = (const float*)d_in[10];
  const float* proj_b   = (const float*)d_in[11];
  const float* ln2_w    = (const float*)d_in[12];
  const float* ln2_b    = (const float*)d_in[13];
  const float* fc1_w    = (const float*)d_in[14];
  const float* fc1_b    = (const float*)d_in[15];
  const float* fc2_w    = (const float*)d_in[16];
  const float* fc2_b    = (const float*)d_in[17];
  const float* normf_w  = (const float*)d_in[18];
  const float* normf_b  = (const float*)d_in[19];
  const float* w_out    = (const float*)d_in[20];
  const float* b_out    = (const float*)d_in[21];
  float* out = (float*)d_out;

  float* ws = (float*)d_ws;
  float* xbuf   = ws;
  float* hbuf   = xbuf   + (size_t)SEQ_N * DIM;
  float* qkvbuf = hbuf   + (size_t)SEQ_N * DIM;
  float* obuf   = qkvbuf + (size_t)SEQ_N * QKV_N;
  float* mbuf   = obuf   + (size_t)SEQ_N * DIM;

  // Embeddings: frames -> rows 16.., text -> rows 0..15
  gemm_kernel<<<dim3(DIM / 128, (F_TOK + 127) / 128), 256, 0, stream>>>(
      x, w_embed, b_embed, nullptr, xbuf + (size_t)N_TEXT * DIM, F_TOK, DIM, EMBED, 0);
  gemm_kernel<<<dim3(DIM / 128, 1), 256, 0, stream>>>(
      text, w_text, b_text, nullptr, xbuf, N_TEXT, DIM, TXT_DIM, 0);

  const int gM = (SEQ_N + 127) / 128;   // 13
  for (int l = 0; l < 4; ++l) {
    ln_kernel<<<SEQ_N, 256, 0, stream>>>(xbuf, ln1_w + l * DIM, ln1_b + l * DIM, hbuf, DIM);
    gemm_kernel<<<dim3(QKV_N / 128, gM), 256, 0, stream>>>(
        hbuf, qkv_w + (size_t)l * DIM * QKV_N, qkv_b + (size_t)l * QKV_N,
        nullptr, qkvbuf, SEQ_N, QKV_N, DIM, 0);
    rope_kernel<<<F_TOK, 256, 0, stream>>>(qkvbuf);
    attn_kernel<<<dim3(99, HEADS), 32, 0, stream>>>(qkvbuf, obuf);
    gemm_kernel<<<dim3(DIM / 128, gM), 256, 0, stream>>>(
        obuf, proj_w + (size_t)l * DIM * DIM, proj_b + (size_t)l * DIM,
        xbuf, xbuf, SEQ_N, DIM, DIM, 0);
    ln_kernel<<<SEQ_N, 256, 0, stream>>>(xbuf, ln2_w + l * DIM, ln2_b + l * DIM, hbuf, DIM);
    gemm_kernel<<<dim3(MLP_H / 128, gM), 256, 0, stream>>>(
        hbuf, fc1_w + (size_t)l * DIM * MLP_H, fc1_b + (size_t)l * MLP_H,
        nullptr, mbuf, SEQ_N, MLP_H, DIM, 1);
    gemm_kernel<<<dim3(DIM / 128, gM), 256, 0, stream>>>(
        mbuf, fc2_w + (size_t)l * MLP_H * DIM, fc2_b + (size_t)l * DIM,
        xbuf, xbuf, SEQ_N, DIM, MLP_H, 0);
  }

  ln_kernel<<<F_TOK, 256, 0, stream>>>(xbuf + (size_t)N_TEXT * DIM, normf_w, normf_b, hbuf, DIM);
  gemm_kernel<<<dim3(EMBED / 128, (F_TOK + 127) / 128), 256, 0, stream>>>(
      hbuf, w_out, b_out, nullptr, out, F_TOK, EMBED, DIM, 0);
}
